// AttentionModel_28355374088790
// MI455X (gfx1250) — matrix operand order repaired and checked by reading, not compiled, not run on hardware
//
#include <hip/hip_runtime.h>
#include <hip/hip_bf16.h>

// ---------- types ----------
typedef __bf16 bf16;
typedef __attribute__((ext_vector_type(16))) __bf16 v16bf;
typedef __attribute__((ext_vector_type(8)))  __bf16 v8bf;
typedef __attribute__((ext_vector_type(8)))  float  v8f;

// sizes
#define BB    128
#define LL    20
#define EE    256
#define HH    512
#define H3    1536
#define H2    1024
#define NCLS  2000
#define CCH   2048
#define NPIX  196
#define MIMG  (BB * NPIX)   // 25088
#define EH    (EE + HH)     // 768

__device__ __forceinline__ bf16 f2bf(float f) {
    union { float f; unsigned int u; } x; x.f = f;
    unsigned int r = x.u + 0x7FFFu + ((x.u >> 16) & 1u);   // round-to-nearest-even
    unsigned short h = (unsigned short)(r >> 16);
    union { unsigned short s; bf16 b; } y; y.s = h;
    return y.b;
}

// ---------- f32 -> bf16 convert ----------
__global__ void k_f32_to_bf16(const float* __restrict__ src, bf16* __restrict__ dst, int n) {
    int i = blockIdx.x * blockDim.x + threadIdx.x;
    if (i < n) dst[i] = f2bf(src[i]);
}

// ---------- image [B,2048,196] -> A_bf16 [B*196, 2048] (transpose + convert) ----------
#define TT 32
__global__ void k_transpose_image(const float* __restrict__ image, bf16* __restrict__ A) {
    __shared__ bf16 tile[TT][TT + 1];
    int b  = blockIdx.x;
    int c0 = blockIdx.y * TT;
    int n0 = blockIdx.z * TT;
    int tx = threadIdx.x;   // 0..31
    int ty = threadIdx.y;   // 0..7
    const float* src = image + (size_t)b * CCH * NPIX;
#pragma unroll
    for (int i = 0; i < TT; i += 8) {
        int c = c0 + ty + i;
        int n = n0 + tx;
        float v = (n < NPIX) ? src[(size_t)c * NPIX + n] : 0.0f;
        tile[ty + i][tx] = f2bf(v);
    }
    __syncthreads();
    bf16* dst = A + (size_t)b * NPIX * CCH;
#pragma unroll
    for (int i = 0; i < TT; i += 8) {
        int n = n0 + ty + i;
        int c = c0 + tx;
        if (n < NPIX) dst[(size_t)n * CCH + c] = tile[tx][ty + i];
    }
}

// ---------- pooled[b,n] = max_c image[b,c,n] ----------
__global__ void k_pool_max(const float* __restrict__ image, float* __restrict__ pooled) {
    int b = blockIdx.x;
    int n = threadIdx.x;
    if (n >= NPIX) return;
    const float* src = image + (size_t)b * CCH * NPIX + n;
    float m = -3.402823466e38f;
    for (int c = 0; c < CCH; ++c) m = fmaxf(m, src[(size_t)c * NPIX]);
    pooled[b * NPIX + n] = m;
}

// ---------- h0 = pooled @ Wimg2h0^T + b  (K=196, tiny -> VALU) ----------
__global__ void k_h0(const float* __restrict__ pooled, const float* __restrict__ W,
                     const float* __restrict__ bias, float* __restrict__ h, bf16* __restrict__ hbf) {
    int i = blockIdx.x * blockDim.x + threadIdx.x;   // b*512 + j
    if (i >= BB * HH) return;
    int b = i >> 9, j = i & 511;
    const float* p = pooled + b * NPIX;
    const float* w = W + (size_t)j * NPIX;
    float acc = bias[j];
    for (int n = 0; n < NPIX; ++n) acc += p[n] * w[n];
    h[i] = acc;
    hbf[i] = f2bf(acc);
}

// ---------- WMMA fragment loads ----------
// A tile (contiguous k order, same as B): lane<16 -> M=lane, K {0..15};
//                                 lane>=16 -> M=lane-16, K {16..31}
__device__ __forceinline__ v16bf load_a16(const bf16* p) {
    v8bf lo = *(const v8bf*)p;          // 16B
    v8bf hi = *(const v8bf*)(p + 8);    // 16B
    return __builtin_shufflevector(lo, hi, 0, 1, 2, 3, 4, 5, 6, 7,
                                           8, 9, 10, 11, 12, 13, 14, 15);
}

__device__ __forceinline__ v8f wmma_bf16(v16bf a, v16bf b, v8f c) {
    return __builtin_amdgcn_wmma_f32_16x16x32_bf16(false, a, false, b, (short)0, c,
                                                   false, false);
}

// ---------- generic bf16 WMMA GEMM: C[M,N] = A[M,K] * W[N,K]^T + bias ----------
// one wave -> 32 x (16*NT) output block; K % 32 == 0, M % 32 == 0, N % 16 == 0
template <int NT>
__global__ void __launch_bounds__(256) k_wmma_gemm(
    const bf16* __restrict__ A, const bf16* __restrict__ W,
    const float* __restrict__ bias, float* __restrict__ C, bf16* __restrict__ Cbf,
    int M, int N, int K, int nStrips, int colStart, int relu) {
    int wave = blockIdx.x * 8 + (threadIdx.x >> 5);
    int lane = threadIdx.x & 31;
    int mStrips = M >> 5;
    if (wave >= mStrips * nStrips) return;           // wave-uniform exit
    int sm = wave / nStrips;
    int sn = wave - sm * nStrips;
    int row0 = sm << 5;                              // 32 rows per wave
    int n0b  = colStart + sn * (16 * NT);

    int half = lane >> 4, mr = lane & 15;
    const bf16* arow0 = A + (size_t)(row0 + mr) * K + half * 16;
    const bf16* arow1 = arow0 + (size_t)16 * K;
    // B tile base: lane holds column n = lane&15, 16 contiguous K at (lane>>4)*16
    const bf16* brow  = W + (size_t)(n0b + mr) * K + half * 16;

    v8f acc[2][NT] = {};
    for (int k0 = 0; k0 < K; k0 += 32) {
        v16bf a0 = load_a16(arow0 + k0);
        v16bf a1 = load_a16(arow1 + k0);
#pragma unroll
        for (int j = 0; j < NT; ++j) {
            v16bf bm = *(const v16bf*)(brow + (size_t)j * 16 * K + k0);  // 32B
            acc[0][j] = wmma_bf16(a0, bm, acc[0][j]);
            acc[1][j] = wmma_bf16(a1, bm, acc[1][j]);
        }
    }

    int ncol = lane & 15;
#pragma unroll
    for (int i = 0; i < 2; ++i) {
#pragma unroll
        for (int j = 0; j < NT; ++j) {
            int n = n0b + j * 16 + ncol;
            float bv = bias ? bias[n] : 0.0f;
#pragma unroll
            for (int v = 0; v < 8; ++v) {
                int m = row0 + i * 16 + v + half * 8;
                float val = acc[i][j][v] + bv;
                if (relu) val = fmaxf(val, 0.0f);
                size_t idx = (size_t)m * N + n;
                C[idx] = val;
                if (Cbf) Cbf[idx] = f2bf(val);
            }
        }
    }
}

// ---------- energy[b,n] = h[b,:] . img_hidden[b,n,:] ----------
__global__ void k_energy(const float* __restrict__ h, const float* __restrict__ ih,
                         float* __restrict__ energy) {
    int i = blockIdx.x * blockDim.x + threadIdx.x;   // b*196+n
    if (i >= BB * NPIX) return;
    int b = i / NPIX;
    const float4* hv = (const float4*)(h + (size_t)b * HH);
    const float4* iv = (const float4*)(ih + (size_t)i * HH);
    float acc = 0.0f;
    for (int j = 0; j < HH / 4; ++j) {
        float4 a = hv[j], c = iv[j];
        acc += a.x * c.x + a.y * c.y + a.z * c.z + a.w * c.w;
    }
    energy[i] = acc;
}

// ---------- softmax over 196, in place, one block per b ----------
__global__ void k_softmax196(float* __restrict__ e) {
    __shared__ float buf[256];
    int b = blockIdx.x, t = threadIdx.x;
    float v = (t < NPIX) ? e[b * NPIX + t] : -3.402823466e38f;
    buf[t] = v;
    __syncthreads();
    for (int s = 128; s > 0; s >>= 1) {
        if (t < s) buf[t] = fmaxf(buf[t], buf[t + s]);
        __syncthreads();
    }
    float mx = buf[0];
    __syncthreads();
    float ev = (t < NPIX) ? __expf(v - mx) : 0.0f;
    buf[t] = ev;
    __syncthreads();
    for (int s = 128; s > 0; s >>= 1) {
        if (t < s) buf[t] += buf[t + s];
        __syncthreads();
    }
    float inv = 1.0f / buf[0];
    if (t < NPIX) e[b * NPIX + t] = ev * inv;
}

// ---------- context[b,j] = sum_n alpha[b,n] * img_hidden[b,n,j] ----------
__global__ void k_context(const float* __restrict__ alpha, const float* __restrict__ ih,
                          float* __restrict__ ctx) {
    int i = blockIdx.x * blockDim.x + threadIdx.x;   // b*512+j
    if (i >= BB * HH) return;
    int b = i >> 9, j = i & 511;
    const float* av = alpha + b * NPIX;
    const float* iv = ih + ((size_t)b * NPIX) * HH + j;
    float acc = 0.0f;
    for (int n = 0; n < NPIX; ++n) acc += av[n] * iv[(size_t)n * HH];
    ctx[i] = acc;
}

// ---------- x = concat(emb[q[:,t]], context) * v  -> bf16 ----------
__global__ void k_xbuild(const int* __restrict__ question, int t, const float* __restrict__ emb,
                         const float* __restrict__ ctx, const float* __restrict__ vvec,
                         bf16* __restrict__ xbf) {
    int i = blockIdx.x * blockDim.x + threadIdx.x;   // b*768+j
    if (i >= BB * EH) return;
    int b = i / EH, j = i - b * EH;
    float val;
    if (j < EE) {
        int q = question[b * LL + t];
        val = emb[(size_t)q * EE + j];
    } else {
        val = ctx[b * HH + (j - EE)];
    }
    xbf[i] = f2bf(val * vvec[j]);
}

// ---------- GRU pointwise (biases already folded into gi/gh by GEMM) ----------
__global__ void k_gru(const float* __restrict__ gi, const float* __restrict__ gh,
                      float* __restrict__ h, bf16* __restrict__ hbf) {
    int i = blockIdx.x * blockDim.x + threadIdx.x;   // b*512+j
    if (i >= BB * HH) return;
    int b = i >> 9, j = i & 511;
    const float* gib = gi + (size_t)b * H3;
    const float* ghb = gh + (size_t)b * H3;
    float ir = gib[j], iz = gib[HH + j], in = gib[2 * HH + j];
    float hr = ghb[j], hz = ghb[HH + j], hn = ghb[2 * HH + j];
    float r = 1.0f / (1.0f + __expf(-(ir + hr)));
    float z = 1.0f / (1.0f + __expf(-(iz + hz)));
    float n = tanhf(in + r * hn);
    float hp = h[i];
    float hv = (1.0f - z) * n + z * hp;
    h[i]  = hv;
    hbf[i] = f2bf(hv);
}

// ---------- host ----------
extern "C" void kernel_launch(void* const* d_in, const int* in_sizes, int n_in,
                              void* d_out, int out_size, void* d_ws, size_t ws_size,
                              hipStream_t stream) {
    const int*   question = (const int*)  d_in[0];
    const float* image    = (const float*)d_in[1];
    const float* emb      = (const float*)d_in[2];
    const float* vvec     = (const float*)d_in[3];
    const float* Wih      = (const float*)d_in[4];
    const float* Whh      = (const float*)d_in[5];
    const float* bih      = (const float*)d_in[6];
    const float* bhh      = (const float*)d_in[7];
    const float* Wimg2h   = (const float*)d_in[8];
    const float* bimg2h   = (const float*)d_in[9];
    const float* Wimg2h0  = (const float*)d_in[10];
    const float* bimg2h0  = (const float*)d_in[11];
    const float* Wfc1     = (const float*)d_in[12];
    const float* bfc1     = (const float*)d_in[13];
    const float* Wfc2     = (const float*)d_in[14];
    const float* bfc2     = (const float*)d_in[15];
    float* out = (float*)d_out;

    char* wsp = (char*)d_ws;
    size_t off = 0;
    auto alloc = [&](size_t bytes) -> void* {
        void* p = wsp + off;
        off += (bytes + 255) & ~(size_t)255;
        return p;
    };

    bf16*  A_bf       = (bf16*) alloc((size_t)MIMG * CCH * 2);
    bf16*  Wimg2h_bf  = (bf16*) alloc((size_t)HH * CCH * 2);
    bf16*  Wih_bf     = (bf16*) alloc((size_t)H3 * EH * 2);
    bf16*  Whh_bf     = (bf16*) alloc((size_t)H3 * HH * 2);
    bf16*  Wfc1_bf    = (bf16*) alloc((size_t)H2 * HH * 2);
    bf16*  Wfc2_bf    = (bf16*) alloc((size_t)NCLS * H2 * 2);
    float* img_hidden = (float*)alloc((size_t)MIMG * HH * 4);
    float* pooled     = (float*)alloc((size_t)BB * NPIX * 4);
    float* hbuf       = (float*)alloc((size_t)BB * HH * 4);
    bf16*  h_bf       = (bf16*) alloc((size_t)BB * HH * 2);
    float* energy     = (float*)alloc((size_t)BB * NPIX * 4);
    float* ctx        = (float*)alloc((size_t)BB * HH * 4);
    bf16*  x_bf       = (bf16*) alloc((size_t)BB * EH * 2);
    float* gi         = (float*)alloc((size_t)BB * H3 * 4);
    float* gh         = (float*)alloc((size_t)BB * H3 * 4);
    float* x1         = (float*)alloc((size_t)BB * H2 * 4);
    bf16*  x1_bf      = (bf16*) alloc((size_t)BB * H2 * 2);
    (void)ws_size; (void)in_sizes; (void)n_in; (void)out_size;

    auto cvt = [&](const float* s, bf16* d, int n) {
        k_f32_to_bf16<<<(n + 255) / 256, 256, 0, stream>>>(s, d, n);
    };
    auto gemm = [&](const bf16* A, const bf16* W, const float* bias, float* C, bf16* Cbf,
                    int M, int N, int K, int relu) {
        int mStrips = M / 32;
        int tilesN  = N / 16;
        int full    = tilesN / 4;          // 64-wide strips
        int tail    = tilesN - full * 4;
        if (full) {
            int waves = mStrips * full;
            k_wmma_gemm<4><<<(waves + 7) / 8, 256, 0, stream>>>(
                A, W, bias, C, Cbf, M, N, K, full, 0, relu);
        }
        if (tail) {
            int waves = mStrips;           // one ragged strip
            int colStart = full * 64;
            switch (tail) {
            case 1:
                k_wmma_gemm<1><<<(waves + 7) / 8, 256, 0, stream>>>(
                    A, W, bias, C, Cbf, M, N, K, 1, colStart, relu);
                break;
            case 2:
                k_wmma_gemm<2><<<(waves + 7) / 8, 256, 0, stream>>>(
                    A, W, bias, C, Cbf, M, N, K, 1, colStart, relu);
                break;
            default:
                k_wmma_gemm<3><<<(waves + 7) / 8, 256, 0, stream>>>(
                    A, W, bias, C, Cbf, M, N, K, 1, colStart, relu);
                break;
            }
        }
    };

    // weight conversions
    cvt(Wimg2h, Wimg2h_bf, HH * CCH);
    cvt(Wih,    Wih_bf,    H3 * EH);
    cvt(Whh,    Whh_bf,    H3 * HH);
    cvt(Wfc1,   Wfc1_bf,   H2 * HH);
    cvt(Wfc2,   Wfc2_bf,   NCLS * H2);

    // image transpose+convert, pooling, h0
    {
        dim3 g(BB, CCH / TT, (NPIX + TT - 1) / TT), b(32, 8);
        k_transpose_image<<<g, b, 0, stream>>>(image, A_bf);
    }
    k_pool_max<<<BB, 256, 0, stream>>>(image, pooled);
    k_h0<<<(BB * HH + 255) / 256, 256, 0, stream>>>(pooled, Wimg2h0, bimg2h0, hbuf, h_bf);

    // img_hidden = A * Wimg2h^T + bimg2h : [25088,2048]x[512,2048] -> [25088,512]
    gemm(A_bf, Wimg2h_bf, bimg2h, img_hidden, nullptr, MIMG, HH, CCH, 0);

    // scan over L=20 time steps
    for (int t = 0; t < LL; ++t) {
        k_energy   <<<(BB * NPIX + 255) / 256, 256, 0, stream>>>(hbuf, img_hidden, energy);
        k_softmax196<<<BB, 256, 0, stream>>>(energy);
        k_context  <<<(BB * HH + 255) / 256, 256, 0, stream>>>(energy, img_hidden, ctx);
        k_xbuild   <<<(BB * EH + 255) / 256, 256, 0, stream>>>(question, t, emb, ctx, vvec, x_bf);
        gemm(x_bf, Wih_bf, bih, gi, nullptr, BB, H3, EH, 0);   // [128,768]x[1536,768]
        gemm(h_bf, Whh_bf, bhh, gh, nullptr, BB, H3, HH, 0);   // [128,512]x[1536,512]
        k_gru      <<<(BB * HH + 255) / 256, 256, 0, stream>>>(gi, gh, hbuf, h_bf);
    }

    // classifier head
    gemm(h_bf,  Wfc1_bf, bfc1, x1,  x1_bf,  BB, H2,   HH, 1);  // relu
    gemm(x1_bf, Wfc2_bf, bfc2, out, nullptr, BB, NCLS, H2, 0);
}
